// MambaMNIST_86912958202461
// MI455X (gfx1250) — compile-verified, matrix operand order checked by reading, not run on hardware
//
#include <hip/hip_runtime.h>
#include <hip/hip_bf16.h>
#include <math.h>

// ---------------------------------------------------------------------------
// Mamba-MNIST forward for gfx1250 (MI455X).
// All dense GEMMs go through v_wmma_f32_16x16x32_f16 (f16 in, f32 accum),
// register double-buffered across K and register-blocked 16x32 per wave.
// Scan / conv / norms stay f32 VALU (tiny, sequential, precision-critical).
// ---------------------------------------------------------------------------

#define DIM       512
#define D_INNER   1024
#define D_STATE   16
#define D_CONV    4
#define DT_RANK   32
#define N_LAYERS  4
#define NUM_CLS   10
#define SEQQ      28
#define BATCH     512
#define NTOK      (BATCH * SEQQ)       // 14336, multiple of 64
#define RMS_EPS   1e-6f

typedef _Float16 v8h  __attribute__((ext_vector_type(8)));
typedef _Float16 v16h __attribute__((ext_vector_type(16)));
typedef float    v8f  __attribute__((ext_vector_type(8)));

// --------------------------- WMMA GEMM --------------------------------------
// C[M,N] = A[M,K] * B[N,K]^T (+bias[n]) (+epilogue)
// A, B row-major f16; K % 32 == 0, M % 64 == 0, N % 64 == 0.
//
// 16-bit A-matrix 16x32 fragment layout (ISA 7.12.2):
//   lanes 0-15  : M = lane,    halves 0..7 -> K=0..7,  halves 8..15 -> K=16..23
//   lanes 16-31 : M = lane-16, halves 0..7 -> K=8..15, halves 8..15 -> K=24..31
// => per lane: two contiguous 8-half (16B) loads at +0/+8 and +16/+24.
// B fragment mirrors A with lane -> N column (B stored as rows of W[N,K]).
// C/D v8f: VGPR vr -> row m0 + vr + 8*(lane>=16), col n0 + (lane&15).
//
// Each wave computes a 16x32 output (one A fragment reused by two B tiles);
// fragments are double-buffered in registers so the v_wmma for slab k issues
// while the global_load_b128s for slab k+32 are still in flight.

enum { EPI_NONE = 0, EPI_SOFTPLUS = 1, EPI_RESIDUAL = 2 };

__device__ __forceinline__ v16h load_frag16(const _Float16* __restrict__ rowp,
                                            int k0, int hi) {
  const _Float16* q = rowp + k0 + hi * 8;
  v8h lo = *reinterpret_cast<const v8h*>(q);
  v8h up = *reinterpret_cast<const v8h*>(q + 16);
  return __builtin_shufflevector(lo, up, 0, 1, 2, 3, 4, 5, 6, 7,
                                         8, 9, 10, 11, 12, 13, 14, 15);
}

template <int EPI>
__global__ __launch_bounds__(256)
void wmma_gemm_kernel(const _Float16* __restrict__ A,   // M x K
                      const _Float16* __restrict__ B,   // N x K
                      const float*    __restrict__ bias,// N or nullptr
                      float*          __restrict__ C,   // M x N (read for residual)
                      int M, int N, int K) {
  const int lane = threadIdx.x & 31;
  const int wave = threadIdx.x >> 5;     // 0..7
  const int r    = lane & 15;
  const int hi   = lane >> 4;            // 0 | 1

  const int tile_m = blockIdx.y * 4 + (wave & 3);   // 16-row tile
  const int npair  = blockIdx.x * 2 + (wave >> 2);  // 32-col pair of tiles
  const int m0 = tile_m * 16;
  const int n0 = npair * 32;
  if (m0 >= M || n0 >= N) return;        // wave-uniform: EXEC stays all-ones

  const _Float16* arow  = A + (size_t)(m0 + r) * K;
  const _Float16* brow0 = B + (size_t)(n0 + r) * K;
  const _Float16* brow1 = B + (size_t)(n0 + 16 + r) * K;

  v8f acc0 = {0.f, 0.f, 0.f, 0.f, 0.f, 0.f, 0.f, 0.f};
  v8f acc1 = {0.f, 0.f, 0.f, 0.f, 0.f, 0.f, 0.f, 0.f};

  // prologue: first K slab
  v16h a  = load_frag16(arow,  0, hi);
  v16h b0 = load_frag16(brow0, 0, hi);
  v16h b1 = load_frag16(brow1, 0, hi);

  int k0 = 0;
  for (; k0 + 32 < K; k0 += 32) {
    if (k0 + 64 < K) {                   // L2 warm-up -> global_prefetch_b8
      __builtin_prefetch(arow + k0 + 64, 0, 1);
      __builtin_prefetch(brow0 + k0 + 64, 0, 1);
      __builtin_prefetch(brow1 + k0 + 64, 0, 1);
    }
    // issue next slab's loads before consuming the current one
    v16h an  = load_frag16(arow,  k0 + 32, hi);
    v16h bn0 = load_frag16(brow0, k0 + 32, hi);
    v16h bn1 = load_frag16(brow1, k0 + 32, hi);
    acc0 = __builtin_amdgcn_wmma_f32_16x16x32_f16(false, a, false, b0,
                                                  (short)0, acc0, false, false);
    acc1 = __builtin_amdgcn_wmma_f32_16x16x32_f16(false, a, false, b1,
                                                  (short)0, acc1, false, false);
    a = an; b0 = bn0; b1 = bn1;
  }
  acc0 = __builtin_amdgcn_wmma_f32_16x16x32_f16(false, a, false, b0,
                                                (short)0, acc0, false, false);
  acc1 = __builtin_amdgcn_wmma_f32_16x16x32_f16(false, a, false, b1,
                                                (short)0, acc1, false, false);

  const int   col0 = n0 + r;
  const int   col1 = n0 + 16 + r;
  const float bv0  = bias ? bias[col0] : 0.0f;
  const float bv1  = bias ? bias[col1] : 0.0f;
#pragma unroll
  for (int vr = 0; vr < 8; ++vr) {
    const int row = m0 + hi * 8 + vr;
    float v0 = acc0[vr] + bv0;
    float v1 = acc1[vr] + bv1;
    if (EPI == EPI_SOFTPLUS) {
      v0 = (v0 > 20.0f) ? v0 : log1pf(__expf(v0));
      v1 = (v1 > 20.0f) ? v1 : log1pf(__expf(v1));
    }
    const size_t i0 = (size_t)row * N + col0;
    const size_t i1 = (size_t)row * N + col1;
    if (EPI == EPI_RESIDUAL) { v0 += C[i0]; v1 += C[i1]; }
    C[i0] = v0;
    C[i1] = v1;
  }
}

// --------------------------- small helper kernels ---------------------------

__global__ __launch_bounds__(256)
void f32_to_f16_kernel(const float* __restrict__ src, _Float16* __restrict__ dst,
                       int n) {
  const int i = blockIdx.x * 256 + threadIdx.x;
  if (i < n) dst[i] = (_Float16)src[i];
}

// pack [rows,28] f32 -> [rows,32] f16, zero-padded (K must be %32 for WMMA)
__global__ __launch_bounds__(256)
void pad28_to_32_kernel(const float* __restrict__ src, _Float16* __restrict__ dst,
                        int nrows) {
  const int i = blockIdx.x * 256 + threadIdx.x;
  const int row = i >> 5, k = i & 31;
  if (row < nrows) dst[i] = (k < 28) ? (_Float16)src[row * 28 + k] : (_Float16)0.0f;
}

// RMSNorm over DIM per token, emit f16 for the following WMMA GEMM
__global__ __launch_bounds__(256)
void rmsnorm_f16_kernel(const float* __restrict__ x, const float* __restrict__ w,
                        _Float16* __restrict__ out) {
  __shared__ float red[256];
  const int t = blockIdx.x;
  const float* xp = x + (size_t)t * DIM;
  float ss = 0.f;
  for (int e = threadIdx.x; e < DIM; e += 256) { float v = xp[e]; ss += v * v; }
  red[threadIdx.x] = ss; __syncthreads();
  for (int s = 128; s > 0; s >>= 1) {
    if (threadIdx.x < s) red[threadIdx.x] += red[threadIdx.x + s];
    __syncthreads();
  }
  const float inv = rsqrtf(red[0] / (float)DIM + RMS_EPS);
  for (int e = threadIdx.x; e < DIM; e += 256)
    out[(size_t)t * DIM + e] = (_Float16)(xp[e] * inv * w[e]);
}

// depthwise causal conv1d (4 taps) + SiLU; input = xz[:, :D_INNER]
__global__ __launch_bounds__(256)
void conv_silu_kernel(const float* __restrict__ xz, const float* __restrict__ cw,
                      const float* __restrict__ cb, float* __restrict__ xc,
                      _Float16* __restrict__ xc16) {
  const int id  = blockIdx.x * 256 + threadIdx.x;   // NTOK * D_INNER threads
  const int d   = id & (D_INNER - 1);
  const int tok = id >> 10;
  const int b   = tok / SEQQ;
  const int t   = tok - b * SEQQ;
  float acc = cb[d];
#pragma unroll
  for (int j = 0; j < D_CONV; ++j) {
    const int tt = t - (D_CONV - 1) + j;
    if (tt >= 0)
      acc += cw[d * D_CONV + j] * xz[((size_t)(b * SEQQ + tt)) * (2 * D_INNER) + d];
  }
  const float s = acc / (1.f + __expf(-acc));
  xc[id]   = s;
  xc16[id] = (_Float16)s;
}

// xdb[:, :32] -> f16 operand for dt_proj GEMM
__global__ __launch_bounds__(256)
void extract_dt_kernel(const float* __restrict__ xdb, _Float16* __restrict__ dt16) {
  const int i = blockIdx.x * 256 + threadIdx.x;     // NTOK*32
  if (i < NTOK * DT_RANK) {
    const int t = i >> 5, j = i & 31;
    dt16[i] = (_Float16)xdb[(size_t)t * (DT_RANK + 2 * D_STATE) + j];
  }
}

// selective scan, fused with y = (y + D*xc)*silu(z), emits f16 for out_proj
__global__ __launch_bounds__(256)
void scan_kernel(const float* __restrict__ xz,    // z = xz[:, D_INNER:]
                 const float* __restrict__ xc,
                 const float* __restrict__ dt,    // softplus already applied
                 const float* __restrict__ xdb,   // B/C at cols 32..63
                 const float* __restrict__ A_log, // [D_INNER, D_STATE] this layer
                 const float* __restrict__ Dp,    // [D_INNER]
                 _Float16* __restrict__ y16) {
  __shared__ float sB[SEQQ][D_STATE];
  __shared__ float sC[SEQQ][D_STATE];
  const int d = blockIdx.x * 256 + threadIdx.x;   // 4 blocks x 256 = D_INNER
  const int b = blockIdx.y;
  for (int i = threadIdx.x; i < SEQQ * 2 * D_STATE; i += 256) {
    const int t = i >> 5, j = i & 31;
    const float v = xdb[((size_t)(b * SEQQ + t)) * (DT_RANK + 2 * D_STATE)
                        + DT_RANK + j];
    if (j < D_STATE) sB[t][j] = v; else sC[t][j - D_STATE] = v;
  }
  __syncthreads();

  float Av[D_STATE];
#pragma unroll
  for (int s = 0; s < D_STATE; ++s) Av[s] = -__expf(A_log[(size_t)d * D_STATE + s]);
  const float Dv = Dp[d];
  float st[D_STATE];
#pragma unroll
  for (int s = 0; s < D_STATE; ++s) st[s] = 0.f;

  for (int t = 0; t < SEQQ; ++t) {
    const size_t tok = (size_t)(b * SEQQ + t);
    const float dtv = dt[tok * D_INNER + d];
    const float xv  = xc[tok * D_INNER + d];
    const float zv  = xz[tok * (2 * D_INNER) + D_INNER + d];
    const float dtx = dtv * xv;
    float y = 0.f;
#pragma unroll
    for (int s = 0; s < D_STATE; ++s) {
      st[s] = st[s] * __expf(dtv * Av[s]) + dtx * sB[t][s];
      y += st[s] * sC[t][s];
    }
    const float sz = zv / (1.f + __expf(-zv));
    y16[tok * D_INNER + d] = (_Float16)((y + Dv * xv) * sz);
  }
}

// final RMSNorm on last token + 10-way classifier
__global__ __launch_bounds__(256)
void head_kernel(const float* __restrict__ h, const float* __restrict__ fw,
                 const float* __restrict__ cw, const float* __restrict__ cb,
                 float* __restrict__ out) {
  __shared__ float red[256];
  __shared__ float nh[DIM];
  const int b = blockIdx.x;
  const float* xp = h + ((size_t)(b * SEQQ + SEQQ - 1)) * DIM;
  float ss = 0.f;
  for (int e = threadIdx.x; e < DIM; e += 256) { float v = xp[e]; ss += v * v; }
  red[threadIdx.x] = ss; __syncthreads();
  for (int s = 128; s > 0; s >>= 1) {
    if (threadIdx.x < s) red[threadIdx.x] += red[threadIdx.x + s];
    __syncthreads();
  }
  const float inv = rsqrtf(red[0] / (float)DIM + RMS_EPS);
  for (int e = threadIdx.x; e < DIM; e += 256) nh[e] = xp[e] * inv * fw[e];
  __syncthreads();
  for (int c = 0; c < NUM_CLS; ++c) {
    float p = 0.f;
    for (int e = threadIdx.x; e < DIM; e += 256) p += nh[e] * cw[c * DIM + e];
    red[threadIdx.x] = p; __syncthreads();
    for (int s = 128; s > 0; s >>= 1) {
      if (threadIdx.x < s) red[threadIdx.x] += red[threadIdx.x + s];
      __syncthreads();
    }
    if (threadIdx.x == 0) out[b * NUM_CLS + c] = red[0] + cb[c];
    __syncthreads();
  }
}

// ------------------------------- launcher -----------------------------------

extern "C" void kernel_launch(void* const* d_in, const int* in_sizes, int n_in,
                              void* d_out, int out_size, void* d_ws, size_t ws_size,
                              hipStream_t stream) {
  (void)in_sizes; (void)n_in; (void)out_size; (void)ws_size;

  const float* x        = (const float*)d_in[0];
  const float* rpw      = (const float*)d_in[1];
  const float* rpb      = (const float*)d_in[2];
  const float* norm_w   = (const float*)d_in[3];
  const float* in_w     = (const float*)d_in[4];
  const float* in_b     = (const float*)d_in[5];
  const float* conv_w   = (const float*)d_in[6];
  const float* conv_b   = (const float*)d_in[7];
  const float* xp_w     = (const float*)d_in[8];
  const float* dtp_w    = (const float*)d_in[9];
  const float* dtp_b    = (const float*)d_in[10];
  const float* A_log    = (const float*)d_in[11];
  const float* Dp       = (const float*)d_in[12];
  const float* out_w    = (const float*)d_in[13];
  const float* out_b    = (const float*)d_in[14];
  const float* fnorm_w  = (const float*)d_in[15];
  const float* cls_w    = (const float*)d_in[16];
  const float* cls_b    = (const float*)d_in[17];
  float* out = (float*)d_out;

  // bump allocator over d_ws (256B aligned slabs)
  char* ws = (char*)d_ws;
  size_t off = 0;
  auto alloc = [&](size_t bytes) -> void* {
    off = (off + 255) & ~(size_t)255;
    void* p = ws + off;
    off += bytes;
    return p;
  };

  float*    h      = (float*)   alloc((size_t)NTOK * DIM * 4);
  _Float16* nh16   = (_Float16*)alloc((size_t)NTOK * DIM * 2);
  float*    xz     = (float*)   alloc((size_t)NTOK * 2 * D_INNER * 4);
  float*    xc     = (float*)   alloc((size_t)NTOK * D_INNER * 4);
  _Float16* xc16   = (_Float16*)alloc((size_t)NTOK * D_INNER * 2);
  float*    xdb    = (float*)   alloc((size_t)NTOK * (DT_RANK + 2 * D_STATE) * 4);
  float*    dtbuf  = (float*)   alloc((size_t)NTOK * D_INNER * 4);
  _Float16* dtin16 = (_Float16*)alloc((size_t)NTOK * DT_RANK * 2);
  _Float16* y16    = (_Float16*)alloc((size_t)NTOK * D_INNER * 2);
  _Float16* rows16 = (_Float16*)alloc((size_t)NTOK * 32 * 2);
  _Float16* roww16 = (_Float16*)alloc((size_t)DIM * 32 * 2);
  _Float16* inw16  = (_Float16*)alloc((size_t)2 * D_INNER * DIM * 2);
  _Float16* xpw16  = (_Float16*)alloc((size_t)(DT_RANK + 2 * D_STATE) * D_INNER * 2);
  _Float16* dtpw16 = (_Float16*)alloc((size_t)D_INNER * DT_RANK * 2);
  _Float16* outw16 = (_Float16*)alloc((size_t)DIM * D_INNER * 2);

  auto cvt = [&](const float* s, _Float16* d, int n) {
    f32_to_f16_kernel<<<(n + 255) / 256, 256, 0, stream>>>(s, d, n);
  };
  // wave: 16x32 out; block (8 waves): 64x64 macro tile
  auto gemm_grid = [](int M, int N) { return dim3(N / 64, M / 64); };

  // ---- row projection: h = pad32(rows) @ pad32(row_proj_w)^T + b ----
  pad28_to_32_kernel<<<(NTOK * 32 + 255) / 256, 256, 0, stream>>>(x, rows16, NTOK);
  pad28_to_32_kernel<<<(DIM * 32 + 255) / 256, 256, 0, stream>>>(rpw, roww16, DIM);
  wmma_gemm_kernel<EPI_NONE><<<gemm_grid(NTOK, DIM), 256, 0, stream>>>(
      rows16, roww16, rpb, h, NTOK, DIM, 32);

  // ---- layers ----
  for (int l = 0; l < N_LAYERS; ++l) {
    cvt(in_w  + (size_t)l * 2 * D_INNER * DIM,  inw16,  2 * D_INNER * DIM);
    cvt(xp_w  + (size_t)l * (DT_RANK + 2 * D_STATE) * D_INNER, xpw16,
        (DT_RANK + 2 * D_STATE) * D_INNER);
    cvt(dtp_w + (size_t)l * D_INNER * DT_RANK,  dtpw16, D_INNER * DT_RANK);
    cvt(out_w + (size_t)l * DIM * D_INNER,      outw16, DIM * D_INNER);

    rmsnorm_f16_kernel<<<NTOK, 256, 0, stream>>>(h, norm_w + (size_t)l * DIM, nh16);

    wmma_gemm_kernel<EPI_NONE><<<gemm_grid(NTOK, 2 * D_INNER), 256, 0, stream>>>(
        nh16, inw16, in_b + (size_t)l * 2 * D_INNER, xz, NTOK, 2 * D_INNER, DIM);

    conv_silu_kernel<<<(NTOK * D_INNER) / 256, 256, 0, stream>>>(
        xz, conv_w + (size_t)l * D_INNER * D_CONV, conv_b + (size_t)l * D_INNER,
        xc, xc16);

    wmma_gemm_kernel<EPI_NONE><<<gemm_grid(NTOK, DT_RANK + 2 * D_STATE), 256, 0,
                                 stream>>>(
        xc16, xpw16, nullptr, xdb, NTOK, DT_RANK + 2 * D_STATE, D_INNER);

    extract_dt_kernel<<<(NTOK * DT_RANK + 255) / 256, 256, 0, stream>>>(xdb, dtin16);

    wmma_gemm_kernel<EPI_SOFTPLUS><<<gemm_grid(NTOK, D_INNER), 256, 0, stream>>>(
        dtin16, dtpw16, dtp_b + (size_t)l * D_INNER, dtbuf, NTOK, D_INNER, DT_RANK);

    scan_kernel<<<dim3(D_INNER / 256, BATCH), 256, 0, stream>>>(
        xz, xc, dtbuf, xdb, A_log + (size_t)l * D_INNER * D_STATE,
        Dp + (size_t)l * D_INNER, y16);

    wmma_gemm_kernel<EPI_RESIDUAL><<<gemm_grid(NTOK, DIM), 256, 0, stream>>>(
        y16, outw16, out_b + (size_t)l * DIM, h, NTOK, DIM, D_INNER);
  }

  // ---- final norm + classifier ----
  head_kernel<<<BATCH, 256, 0, stream>>>(h, fnorm_w, cls_w, cls_b, out);
}
